// Pairwise_Pointer_Network_44229573214681
// MI455X (gfx1250) — matrix operand order repaired and checked by reading, not compiled, not run on hardware
//
#include <hip/hip_runtime.h>
#include <math.h>

typedef __bf16 bf16;
typedef __bf16 v16bf __attribute__((ext_vector_type(16)));
typedef __bf16 v8bf  __attribute__((ext_vector_type(8)));
typedef float  v8f   __attribute__((ext_vector_type(8)));

union BF16Frag { v16bf v; v8bf h[2]; };

static __device__ __forceinline__ unsigned short f2bf_bits(float f) {
  union { float f; unsigned u; } x; x.f = f;
  unsigned r = x.u + 0x7FFFu + ((x.u >> 16) & 1u);
  return (unsigned short)(r >> 16);
}

static __device__ __forceinline__ int imin(int a, int b) { return a < b ? a : b; }

// ---------------------------------------------------------------------------
// Conversion / layout kernels
// ---------------------------------------------------------------------------
__global__ __launch_bounds__(256)
void k_f32_to_bf16(const float4* __restrict__ src, unsigned short* __restrict__ dst,
                   long long n4) {
  long long i = (long long)blockIdx.x * 256 + threadIdx.x;
  if (i >= n4) return;
  float4 v = src[i];
  ushort4 o;
  o.x = f2bf_bits(v.x); o.y = f2bf_bits(v.y);
  o.z = f2bf_bits(v.z); o.w = f2bf_bits(v.w);
  *(ushort4*)(dst + 4 * i) = o;
}

// dst[n*dstK + k] = (n<validN && k<validK) ? src[k*src_ld + src_col0 + n] : 0
__global__ __launch_bounds__(256)
void k_transpose_bf16(const float* __restrict__ src, int src_ld, int src_col0,
                      unsigned short* __restrict__ dst, int dstK, int dstN,
                      int validN, int validK) {
  long long i = (long long)blockIdx.x * 256 + threadIdx.x;
  long long total = (long long)dstN * dstK;
  if (i >= total) return;
  int n = (int)(i / dstK);
  int k = (int)(i % dstK);
  float v = (n < validN && k < validK) ? src[(long long)k * src_ld + src_col0 + n] : 0.0f;
  dst[i] = f2bf_bits(v);
}

// dst[r*dstK + k] = (k<validK) ? src[r*src_ld + k] : 0   (pad K, no transpose)
__global__ __launch_bounds__(256)
void k_pad_bf16(const float* __restrict__ src, int src_ld,
                unsigned short* __restrict__ dst, int dstK, int rows, int validK) {
  long long i = (long long)blockIdx.x * 256 + threadIdx.x;
  long long total = (long long)rows * dstK;
  if (i >= total) return;
  int r = (int)(i / dstK);
  int k = (int)(i % dstK);
  float v = (k < validK) ? src[(long long)r * src_ld + k] : 0.0f;
  dst[i] = f2bf_bits(v);
}

// ---------------------------------------------------------------------------
// Generic WMMA GEMM:  C[M,N] = act(A[M,K] * Bt[N,K]^T + bias)
// A, Bt row-major bf16; C row-major bf16 or f32.  K % 32 == 0.
// 256 threads = 8 waves (2 x 4); block tile 128x256, wave tile 64x64.
// Double-buffered LDS, one barrier per K-step, unpredicated (clamped) loads.
// ---------------------------------------------------------------------------
#define BM 128
#define BN 256
#define BK 32
#define LDSK 40   // padded LDS pitch (bf16) to spread banks

template<int ACT, int OUT_BF16>   // ACT: 0 none, 1 relu, 2 tanh
__global__ __launch_bounds__(256)
void gemm_kernel(const bf16* __restrict__ A, int lda, long long strideA,
                 const bf16* __restrict__ Bt, int ldb, long long strideB,
                 void* __restrict__ Cout, int ldc, long long strideC,
                 int M, int N, int K,
                 const float* __restrict__ bias, float bias_scalar) {
  __shared__ __align__(16) bf16 As[2][BM * LDSK];
  __shared__ __align__(16) bf16 Bs[2][BN * LDSK];

  const int bz = blockIdx.z;
  A  += strideA * bz;
  Bt += strideB * bz;

  const int m0 = blockIdx.y * BM;
  const int n0 = blockIdx.x * BN;

  const int tid  = threadIdx.x;
  const int lane = tid & 31;
  const int wave = tid >> 5;
  const int wm   = wave >> 2;   // 0..1
  const int wn   = wave & 3;    // 0..3

  v8f acc[4][4];
  #pragma unroll
  for (int i = 0; i < 4; ++i)
    #pragma unroll
    for (int j = 0; j < 4; ++j)
      #pragma unroll
      for (int r = 0; r < 8; ++r) acc[i][j][r] = 0.0f;

  // global->LDS mapping: rows clamped (out-of-range tile rows/cols only feed
  // masked-out C elements, so duplicated data is harmless and loads never fault)
  const int lrow = tid >> 2;          // 0..63
  const int lk   = (tid & 3) * 8;     // 0,8,16,24

  const bf16* aP[2];
  #pragma unroll
  for (int s = 0; s < 2; ++s)
    aP[s] = A + (long long)imin(m0 + lrow + s * 64, M - 1) * lda + lk;
  const bf16* bP[4];
  #pragma unroll
  for (int s = 0; s < 4; ++s)
    bP[s] = Bt + (long long)imin(n0 + lrow + s * 64, N - 1) * ldb + lk;

  // fragment lane mapping
  const int frow = lane & 15;
  const int kbA  = (lane >> 4) * 8;   // A: K {0..7,16..23} / {8..15,24..31}
  const int kbB  = (lane >> 4) * 8;   // B: K {0..7,16..23} / {8..15,24..31}

  const int nk = K / BK;

  v8bf areg[2], breg[4];
  // prologue: tile 0
  #pragma unroll
  for (int s = 0; s < 2; ++s) areg[s] = *(const v8bf*)(aP[s]);
  #pragma unroll
  for (int s = 0; s < 4; ++s) breg[s] = *(const v8bf*)(bP[s]);
  #pragma unroll
  for (int s = 0; s < 2; ++s) *(v8bf*)(&As[0][(lrow + s * 64) * LDSK + lk]) = areg[s];
  #pragma unroll
  for (int s = 0; s < 4; ++s) *(v8bf*)(&Bs[0][(lrow + s * 64) * LDSK + lk]) = breg[s];
  __syncthreads();

  for (int t = 0; t < nk; ++t) {
    const int cur = t & 1;
    const int nxt = cur ^ 1;

    // issue next tile's global loads (latency hidden behind WMMAs below)
    if (t + 1 < nk) {
      const int ko = (t + 1) * BK;
      #pragma unroll
      for (int s = 0; s < 2; ++s) areg[s] = *(const v8bf*)(aP[s] + ko);
      #pragma unroll
      for (int s = 0; s < 4; ++s) breg[s] = *(const v8bf*)(bP[s] + ko);
    }

    // compute on current buffer
    BF16Frag af[4];
    #pragma unroll
    for (int i = 0; i < 4; ++i) {
      const bf16* p = &As[cur][(wm * 64 + i * 16 + frow) * LDSK];
      af[i].h[0] = *(const v8bf*)(p + kbA);
      af[i].h[1] = *(const v8bf*)(p + kbA + 16);
    }
    #pragma unroll
    for (int j = 0; j < 4; ++j) {
      BF16Frag bfr;
      const bf16* p = &Bs[cur][(wn * 64 + j * 16 + frow) * LDSK];
      bfr.h[0] = *(const v8bf*)(p + kbB);
      bfr.h[1] = *(const v8bf*)(p + kbB + 16);
      #pragma unroll
      for (int i = 0; i < 4; ++i)
        acc[i][j] = __builtin_amdgcn_wmma_f32_16x16x32_bf16(
            false, af[i].v, false, bfr.v, (short)0, acc[i][j], false, false);
    }

    if (t + 1 < nk) {
      #pragma unroll
      for (int s = 0; s < 2; ++s) *(v8bf*)(&As[nxt][(lrow + s * 64) * LDSK + lk]) = areg[s];
      #pragma unroll
      for (int s = 0; s < 4; ++s) *(v8bf*)(&Bs[nxt][(lrow + s * 64) * LDSK + lk]) = breg[s];
      __syncthreads();
    }
  }

  // epilogue: C/D layout -> lanes 0-15: M=r, N=lane; lanes 16-31: M=8+r, N=lane-16
  const int cn = lane & 15;
  const int rb = (lane >> 4) * 8;
  #pragma unroll
  for (int i = 0; i < 4; ++i) {
    #pragma unroll
    for (int j = 0; j < 4; ++j) {
      int gc = n0 + wn * 64 + j * 16 + cn;
      if (gc < N) {
        float bv = bias ? bias[gc] : bias_scalar;
        #pragma unroll
        for (int r = 0; r < 8; ++r) {
          int gr = m0 + wm * 64 + i * 16 + rb + r;
          if (gr < M) {
            float v = acc[i][j][r] + bv;
            if (ACT == 1) v = fmaxf(v, 0.0f);
            if (ACT == 2) v = tanhf(v);
            long long idx = strideC * bz + (long long)gr * ldc + gc;
            if (OUT_BF16) ((unsigned short*)Cout)[idx] = f2bf_bits(v);
            else          ((float*)Cout)[idx] = v;
          }
        }
      }
    }
  }
}

// ---------------------------------------------------------------------------
// Mask + term2 gather + softmax: one wave per (b, t) row, t = 1..199
// logits[b,t,h] = (h<t) ? term1[b,t,h] + Q[b,h,t-h] + bil_b : 0 ; softmax over h
// ---------------------------------------------------------------------------
__global__ __launch_bounds__(256)
void k_softmax(const float* __restrict__ term1,  // [256,200,208]
               const float* __restrict__ Q,      // [256*200,208]
               const float* __restrict__ bil_b,
               float* __restrict__ out) {        // [256,199,200]
  const int row  = blockIdx.x * 8 + (threadIdx.x >> 5);
  const int lane = threadIdx.x & 31;
  if (row >= 256 * 199) return;
  const int b = row / 199;
  const int t = row % 199 + 1;
  const float bb = *bil_b;
  const float* trow = term1 + ((long long)b * 200 + t) * 208;

  float vals[7];
  float mx = 0.0f;  // masked zeros always present in the row
  #pragma unroll
  for (int it = 0; it < 7; ++it) {
    int h = it * 32 + lane;
    float v = -1e30f;
    if (h < 200) {
      v = 0.0f;
      if (h < t) v = trow[h] + Q[((long long)b * 200 + h) * 208 + (t - h)] + bb;
    }
    vals[it] = v;
    mx = fmaxf(mx, v);
  }
  #pragma unroll
  for (int off = 16; off > 0; off >>= 1) mx = fmaxf(mx, __shfl_xor(mx, off, 32));
  float s = 0.0f;
  #pragma unroll
  for (int it = 0; it < 7; ++it) {
    int h = it * 32 + lane;
    float e = (h < 200) ? __expf(vals[it] - mx) : 0.0f;
    vals[it] = e;
    s += e;
  }
  #pragma unroll
  for (int off = 16; off > 0; off >>= 1) s += __shfl_xor(s, off, 32);
  float inv = 1.0f / s;
  #pragma unroll
  for (int it = 0; it < 7; ++it) {
    int h = it * 32 + lane;
    if (h < 200) out[((long long)b * 199 + (t - 1)) * 200 + h] = vals[it] * inv;
  }
}

// ---------------------------------------------------------------------------
extern "C" void kernel_launch(void* const* d_in, const int* in_sizes, int n_in,
                              void* d_out, int out_size, void* d_ws, size_t ws_size,
                              hipStream_t stream) {
  const float* paragraphs = (const float*)d_in[0];
  const float* head_W1 = (const float*)d_in[1];
  const float* head_b1 = (const float*)d_in[2];
  const float* head_W2 = (const float*)d_in[3];
  const float* head_b2 = (const float*)d_in[4];
  const float* tail_W1 = (const float*)d_in[5];
  const float* tail_b1 = (const float*)d_in[6];
  const float* tail_W2 = (const float*)d_in[7];
  const float* tail_b2 = (const float*)d_in[8];
  const float* rel_emb = (const float*)d_in[9];
  const float* bil_W   = (const float*)d_in[10];
  const float* bil_b   = (const float*)d_in[11];
  float* out = (float*)d_out;

  const long long ROWS = 51200;  // B*N

  char* ws = (char*)d_ws;
  size_t off = 0;
  auto alloc = [&](size_t bytes) -> void* {
    void* p = ws + off;
    off += (bytes + 255) & ~(size_t)255;
    return p;
  };
  bf16*  Xb    = (bf16*)alloc(ROWS * 512 * 2);
  bf16*  hid   = (bf16*)alloc(ROWS * 512 * 2);
  bf16*  headb = (bf16*)alloc(ROWS * 512 * 2);
  bf16*  tailb = (bf16*)alloc(ROWS * 512 * 2);
  bf16*  hWb   = (bf16*)alloc(ROWS * 512 * 2);
  bf16*  Pb    = (bf16*)alloc(ROWS * 224 * 2);
  float* Qf    = (float*)alloc(ROWS * 208 * 4);
  float* T1    = (float*)alloc(256LL * 200 * 208 * 4);
  bf16*  W1hT  = (bf16*)alloc(512 * 512 * 2);
  bf16*  W2hT  = (bf16*)alloc(512 * 512 * 2);
  bf16*  W1tT  = (bf16*)alloc(512 * 512 * 2);
  bf16*  W2tT  = (bf16*)alloc(512 * 512 * 2);
  bf16*  WtlT  = (bf16*)alloc(512 * 512 * 2);
  bf16*  WrlT  = (bf16*)alloc(224 * 512 * 2);
  bf16*  RelT  = (bf16*)alloc(200 * 224 * 2);

  // ---- convert activations + weights to bf16 (weights transposed to N x K) --
  long long nX4 = ROWS * 512 / 4;
  k_f32_to_bf16<<<(unsigned)((nX4 + 255) / 256), 256, 0, stream>>>(
      (const float4*)paragraphs, (unsigned short*)Xb, nX4);

  unsigned gW = (unsigned)((512LL * 512 + 255) / 256);
  k_transpose_bf16<<<gW, 256, 0, stream>>>(head_W1, 512, 0, (unsigned short*)W1hT, 512, 512, 512, 512);
  k_transpose_bf16<<<gW, 256, 0, stream>>>(head_W2, 512, 0, (unsigned short*)W2hT, 512, 512, 512, 512);
  k_transpose_bf16<<<gW, 256, 0, stream>>>(tail_W1, 512, 0, (unsigned short*)W1tT, 512, 512, 512, 512);
  k_transpose_bf16<<<gW, 256, 0, stream>>>(tail_W2, 512, 0, (unsigned short*)W2tT, 512, 512, 512, 512);
  k_transpose_bf16<<<gW, 256, 0, stream>>>(bil_W, 712, 0,   (unsigned short*)WtlT, 512, 512, 512, 512);
  k_transpose_bf16<<<(unsigned)((224LL * 512 + 255) / 256), 256, 0, stream>>>(
      bil_W, 712, 512, (unsigned short*)WrlT, 512, 224, 200, 512);
  k_pad_bf16<<<(unsigned)((200LL * 224 + 255) / 256), 256, 0, stream>>>(
      rel_emb, 200, (unsigned short*)RelT, 224, 200, 200);

  // ---- MLPs (51200 x 512 x 512 GEMMs), block tile 128x256 ------------------
  dim3 gBig(2, 400, 1);
  gemm_kernel<1, 1><<<gBig, 256, 0, stream>>>(Xb, 512, 0, W1hT, 512, 0, hid,   512, 0, (int)ROWS, 512, 512, head_b1, 0.f);
  gemm_kernel<2, 1><<<gBig, 256, 0, stream>>>(hid, 512, 0, W2hT, 512, 0, headb, 512, 0, (int)ROWS, 512, 512, head_b2, 0.f);
  gemm_kernel<1, 1><<<gBig, 256, 0, stream>>>(Xb, 512, 0, W1tT, 512, 0, hid,   512, 0, (int)ROWS, 512, 512, tail_b1, 0.f);
  gemm_kernel<2, 1><<<gBig, 256, 0, stream>>>(hid, 512, 0, W2tT, 512, 0, tailb, 512, 0, (int)ROWS, 512, 512, tail_b2, 0.f);

  // ---- hW = head * W_tail ; P = head * W_rel (N padded 224) ----------------
  gemm_kernel<0, 1><<<gBig, 256, 0, stream>>>(headb, 512, 0, WtlT, 512, 0, hWb, 512, 0, (int)ROWS, 512, 512, nullptr, 0.f);
  gemm_kernel<0, 1><<<dim3(1, 400, 1), 256, 0, stream>>>(headb, 512, 0, WrlT, 512, 0, Pb, 224, 0, (int)ROWS, 224, 512, nullptr, 0.f);

  // ---- Q = P * rel_emb^T  (K padded 224, f32 out) --------------------------
  gemm_kernel<0, 0><<<dim3(1, 400, 1), 256, 0, stream>>>(Pb, 224, 0, RelT, 224, 0, Qf, 208, 0, (int)ROWS, 200, 224, nullptr, 0.f);

  // ---- term1: batched 200x512x200, C = tail_b * hW_b^T ---------------------
  gemm_kernel<0, 0><<<dim3(1, 2, 256), 256, 0, stream>>>(
      tailb, 512, 200LL * 512, hWb, 512, 200LL * 512, T1, 208, 200LL * 208,
      200, 200, 512, nullptr, 0.f);

  // ---- mask + gather term2 + softmax ---------------------------------------
  k_softmax<<<6368, 256, 0, stream>>>(T1, Qf, bil_b, out);
}
